// SingleHead_30348238913603
// MI455X (gfx1250) — compile-verified, matrix operand order checked by reading, not run on hardware
//
#include <hip/hip_runtime.h>
#include <hip/hip_bf16.h>

// Problem dims (from reference): B=4, T=4096, E=1024, HD=128
#define BB 4
#define TT 4096
#define EE 1024
#define HD 128

typedef _Float16 h8   __attribute__((ext_vector_type(8)));
typedef _Float16 v16h __attribute__((ext_vector_type(16)));
typedef float    v8f  __attribute__((ext_vector_type(8)));
typedef int      v4i  __attribute__((vector_size(16)));   // matches builtin param

#if defined(__AMDGCN__) && __has_builtin(__builtin_amdgcn_global_load_async_to_lds_b128)
#define USE_ASYNC_LDS 1
typedef __attribute__((address_space(1))) v4i* gas_v4i_ptr;
typedef __attribute__((address_space(3))) v4i* las_v4i_ptr;
#else
#define USE_ASYNC_LDS 0
#endif

// Build a 16x32 f16 WMMA fragment (A layout; B uses the same per-lane packing
// with lane meaning the N column, loaded column-major) from a row-major
// 32-half segment. Per ISA 7.12.2:
//   lanes 0-15 : K elements {0..7, 16..23}
//   lanes 16-31: K elements {8..15, 24..31}
// => two contiguous 16B loads per lane.
static __device__ inline v16h ld_frag_h(const _Float16* seg, int hi) {
  const h8* p = (const h8*)seg;
  h8 a = p[hi ? 1 : 0];   // halves 8..15  or 0..7
  h8 b = p[hi ? 3 : 2];   // halves 24..31 or 16..23
  return __builtin_shufflevector(a, b, 0,1,2,3,4,5,6,7,8,9,10,11,12,13,14,15);
}

static __device__ inline v16h frag_from_pair(h8 a, h8 b) {
  return __builtin_shufflevector(a, b, 0,1,2,3,4,5,6,7,8,9,10,11,12,13,14,15);
}

// ---------------------------------------------------------------------------
// Kernel 0: pre-swizzle Wq/Wk/Wv (f32 [E][HD]) into fragment-contiguous f16.
// For matrix m, k-chunk c (32 deep), n-tile nt (16 wide), lane l: the 16
// halves that lane l of a wave needs for the B fragment are stored
// contiguously at  m*128K + (((c*8 + nt)*32) + l)*16.  One-time 1.5 MB read;
// turns the projection's 48 strided scalar loads/iter into 6 b128 loads.
// ---------------------------------------------------------------------------
__global__ __launch_bounds__(256) void pack_w_kernel(
    const float* __restrict__ Wq, const float* __restrict__ Wk,
    const float* __restrict__ Wv, _Float16* __restrict__ wsw)
{
  int t = blockIdx.x * 256 + threadIdx.x;       // [0, 3*32*8*32)
  int m    = t >> 13;                           // matrix 0..2
  int rem  = t & 8191;
  int c    = rem >> 8;                          // k-chunk 0..31
  int nt   = (rem >> 5) & 7;                    // n-tile 0..7
  int lane = rem & 31;
  int hi   = (lane >> 4) & 1;
  int col  = lane & 15;
  const float* W = (m == 0) ? Wq : (m == 1) ? Wk : Wv;
  int n = nt * 16 + col;
  h8 lo8, up8;
#pragma unroll
  for (int e = 0; e < 8; ++e) {
    int ka = c * 32 + (hi ? 8 : 0) + e;
    lo8[e] = (_Float16)W[ka * HD + n];
    up8[e] = (_Float16)W[(ka + 16) * HD + n];
  }
  _Float16* dst = wsw + (size_t)m * (32 * 8 * 32 * 16) + (size_t)t % 8192 * 16;
  *(h8*)dst = lo8;
  *(h8*)(dst + 8) = up8;
}

// ---------------------------------------------------------------------------
// Kernel 1: fused QKV projection.  Y = X @ W{q,k,v},  X:[B*T,E] f32.
// Grid: (B*T)/16 blocks of 256 threads (8 waves). Block owns one 16-row
// M-tile; wave w owns N-tile columns [16w,16w+16). The 16x32 x tile is
// staged (f32->f16) through LDS once per k-chunk and shared by all 8 waves;
// B fragments come from the pre-swizzled W as two contiguous b128 loads each.
// ---------------------------------------------------------------------------
__global__ __launch_bounds__(256) void qkv_proj_kernel(
    const float* __restrict__ x, const _Float16* __restrict__ wsw,
    _Float16* __restrict__ qh, _Float16* __restrict__ kh,
    _Float16* __restrict__ vh)
{
  __shared__ __align__(16) _Float16 Xs[16 * 32];   // 1 KB staged x tile

  const int tid  = threadIdx.x;
  const int lane = tid & 31;
  const int wave = tid >> 5;     // 0..7 -> N tile
  const int hi   = (lane >> 4) & 1;
  const int col  = lane & 15;
  const int m0   = blockIdx.x * 16;

  const _Float16* wq = wsw;
  const _Float16* wk = wsw + 32 * 8 * 32 * 16;
  const _Float16* wv = wk + 32 * 8 * 32 * 16;
  const size_t fragoff = ((size_t)wave * 32 + lane) * 16;  // + c*8*32*16

  v8f accq = {}, acck = {}, accv = {};

  for (int c = 0; c < EE / 32; ++c) {
    // stage x[m0..m0+16, c*32..c*32+32) -> f16 LDS (coalesced, read once)
#pragma unroll
    for (int i = tid; i < 512; i += 256) {
      int row = i >> 5, kc = i & 31;
      Xs[i] = (_Float16)x[((size_t)m0 + row) * EE + c * 32 + kc];
    }
    __syncthreads();

    v16h af = ld_frag_h(&Xs[col * 32], hi);

    const size_t fo = (size_t)c * (8 * 32 * 16) + fragoff;
    h8 q0 = *(const h8*)(wq + fo), q1 = *(const h8*)(wq + fo + 8);
    h8 k0 = *(const h8*)(wk + fo), k1 = *(const h8*)(wk + fo + 8);
    h8 v0 = *(const h8*)(wv + fo), v1 = *(const h8*)(wv + fo + 8);
    v16h bq = frag_from_pair(q0, q1);
    v16h bk = frag_from_pair(k0, k1);
    v16h bv = frag_from_pair(v0, v1);

    accq = __builtin_amdgcn_wmma_f32_16x16x32_f16(false, af, false, bq, (short)0, accq, false, false);
    acck = __builtin_amdgcn_wmma_f32_16x16x32_f16(false, af, false, bk, (short)0, acck, false, false);
    accv = __builtin_amdgcn_wmma_f32_16x16x32_f16(false, af, false, bv, (short)0, accv, false, false);
    __syncthreads();
  }

  // C/D layout: lane group gives N=col; VGPR r -> row r (+8 for hi lanes)
#pragma unroll
  for (int r = 0; r < 8; ++r) {
    size_t row = (size_t)m0 + r + (hi ? 8 : 0);
    size_t idx = row * HD + wave * 16 + col;
    qh[idx] = (_Float16)accq[r];
    kh[idx] = (_Float16)acck[r];
    vh[idx] = (_Float16)accv[r];
  }
}

// ---------------------------------------------------------------------------
// Kernel 2: causal flash attention, one (batch, 64-query block) per 128-thread
// workgroup (4 waves; wave w owns query rows qw..qw+15). Key blocks of 32
// staged into LDS (K via async global->LDS when available, V transposed so PV
// B-fragments are contiguous), online softmax with 16-lane butterfly
// reductions, P reshaped C->A layout through wave-private LDS.
// ---------------------------------------------------------------------------
__global__ __launch_bounds__(128) void flash_attn_kernel(
    const _Float16* __restrict__ qh, const _Float16* __restrict__ kh,
    const _Float16* __restrict__ vh, float* __restrict__ out)
{
  __shared__ __align__(16) _Float16 Kt[32 * HD];      // [key][d]   8 KB
  __shared__ __align__(16) _Float16 Vt[HD * 32];      // [d][key]   8 KB
  __shared__ __align__(16) _Float16 Pb[4][16 * 32];   // per-wave P 4 KB

  const int tid  = threadIdx.x;
  const int lane = tid & 31;
  const int wave = tid >> 5;           // 0..3 query sub-tile
  const int hi   = (lane >> 4) & 1;
  const int col  = lane & 15;
  const int b    = blockIdx.y;
  const int q0   = blockIdx.x * 64;
  const int qw   = q0 + wave * 16;
  const size_t rowbase = (size_t)b * TT;

  // Q fragments: 4 chunks of 32 along head dim (f16, 16B loads)
  v16h qf[4];
  {
    const _Float16* qp = qh + (rowbase + qw + col) * HD;
#pragma unroll
    for (int c = 0; c < 4; ++c) qf[c] = ld_frag_h(qp + c * 32, hi);
  }

  v8f zero = {};
  v8f o[8];
#pragma unroll
  for (int t = 0; t < 8; ++t) o[t] = zero;
  float mrow[8], lrow[8];
#pragma unroll
  for (int r = 0; r < 8; ++r) { mrow[r] = -__builtin_inff(); lrow[r] = 0.f; }

  const float sc  = 0.08838834764831845f;   // 1/sqrt(128)
  const int   nkb = blockIdx.x * 2 + 2;     // key blocks covering q0..q0+63

  for (int kb = 0; kb < nkb; ++kb) {
    const int k0 = kb * 32;
    // ---- stage K (async copy if available) and V (transpose) into LDS ----
    {
      const h8* ks = (const h8*)(kh + (rowbase + k0) * HD);
#if USE_ASYNC_LDS
      for (int i = tid; i < 512; i += 128)
        __builtin_amdgcn_global_load_async_to_lds_b128(
            (gas_v4i_ptr)(ks + i), (las_v4i_ptr)&Kt[i * 8], 0, 0);
#else
      h8* kd = (h8*)Kt;
      for (int i = tid; i < 512; i += 128) kd[i] = ks[i];
#endif
      const h8* vs = (const h8*)(vh + (rowbase + k0) * HD);
      for (int i = tid; i < 512; i += 128) {
        int key = i >> 4, d0 = (i & 15) * 8;
        h8 v = vs[i];
#pragma unroll
        for (int e = 0; e < 8; ++e) Vt[(d0 + e) * 32 + key] = v[e];
      }
      // prefetch next key block's K rows (global_prefetch_b8)
      if (kb + 1 < nkb)
        __builtin_prefetch(kh + (rowbase + k0 + 32) * HD + tid * 32, 0, 1);
#if USE_ASYNC_LDS
      asm volatile("s_wait_asynccnt 0x0" ::: "memory");
#endif
    }
    __syncthreads();

    if (k0 <= qw + 15) {   // wave-uniform causal early-out (barriers stay uniform)
      // ---- S = Q K^T : two 16x16 tiles over the 32-key block ----
      v8f s0 = {}, s1 = {};
#pragma unroll
      for (int c = 0; c < 4; ++c) {
        v16h b0 = ld_frag_h(&Kt[col * HD + c * 32], hi);
        v16h b1 = ld_frag_h(&Kt[(col + 16) * HD + c * 32], hi);
        s0 = __builtin_amdgcn_wmma_f32_16x16x32_f16(false, qf[c], false, b0, (short)0, s0, false, false);
        s1 = __builtin_amdgcn_wmma_f32_16x16x32_f16(false, qf[c], false, b1, (short)0, s1, false, false);
      }
      // ---- online softmax (row = one VGPR across a 16-lane half) ----
      float p0[8], p1[8], corr[8];
#pragma unroll
      for (int r = 0; r < 8; ++r) {
        int q = qw + r + (hi ? 8 : 0);
        float a0 = s0[r] * sc; if (k0 + col      > q) a0 = -__builtin_inff();
        float a1 = s1[r] * sc; if (k0 + 16 + col > q) a1 = -__builtin_inff();
        float mx = fmaxf(a0, a1);
#pragma unroll
        for (int off = 1; off < 16; off <<= 1) mx = fmaxf(mx, __shfl_xor(mx, off, 32));
        float mn = fmaxf(mrow[r], mx);
        float e0 = __expf(a0 - mn);
        float e1 = __expf(a1 - mn);
        float su = e0 + e1;
#pragma unroll
        for (int off = 1; off < 16; off <<= 1) su += __shfl_xor(su, off, 32);
        corr[r] = __expf(mrow[r] - mn);
        mrow[r] = mn;
        lrow[r] = lrow[r] * corr[r] + su;
        p0[r] = e0; p1[r] = e1;
      }
      // ---- reshape P (C layout) -> A fragment via wave-private LDS ----
      _Float16* pb = &Pb[wave][0];
#pragma unroll
      for (int r = 0; r < 8; ++r) {
        int row = r + (hi ? 8 : 0);
        pb[row * 32 + col]      = (_Float16)p0[r];
        pb[row * 32 + 16 + col] = (_Float16)p1[r];
      }
      // in-wave cross-lane LDS RAW: drain DS counter explicitly (CDNA5 split wait)
      asm volatile("s_wait_dscnt 0x0" ::: "memory");
      v16h pf = ld_frag_h(pb + col * 32, hi);
      // ---- O = diag(corr)·O + P·V  (8 head-dim tiles) ----
#pragma unroll
      for (int t = 0; t < 8; ++t) {
#pragma unroll
        for (int r = 0; r < 8; ++r) o[t][r] *= corr[r];
        v16h bv = ld_frag_h(&Vt[(t * 16 + col) * 32], hi);
        o[t] = __builtin_amdgcn_wmma_f32_16x16x32_f16(false, pf, false, bv, (short)0, o[t], false, false);
      }
    }
    __syncthreads();
  }

  // ---- normalize and store f32 output ----
#pragma unroll
  for (int r = 0; r < 8; ++r) {
    float inv = 1.0f / lrow[r];
    size_t row = rowbase + qw + r + (hi ? 8 : 0);
#pragma unroll
    for (int t = 0; t < 8; ++t)
      out[row * HD + t * 16 + col] = o[t][r] * inv;
  }
}

// ---------------------------------------------------------------------------
extern "C" void kernel_launch(void* const* d_in, const int* in_sizes, int n_in,
                              void* d_out, int out_size, void* d_ws, size_t ws_size,
                              hipStream_t stream) {
  (void)in_sizes; (void)n_in; (void)out_size; (void)ws_size;
  const float* x  = (const float*)d_in[0];
  const float* Wq = (const float*)d_in[1];
  const float* Wk = (const float*)d_in[2];
  const float* Wv = (const float*)d_in[3];
  float* out = (float*)d_out;

  // workspace: f16 Q,K,V (12 MB) + swizzled f16 W (768 KB)
  _Float16* qh  = (_Float16*)d_ws;
  _Float16* kh  = qh + (size_t)BB * TT * HD;
  _Float16* vh  = kh + (size_t)BB * TT * HD;
  _Float16* wsw = vh + (size_t)BB * TT * HD;

  pack_w_kernel<<<dim3(96), dim3(256), 0, stream>>>(Wq, Wk, Wv, wsw);
  qkv_proj_kernel<<<dim3((BB * TT) / 16), dim3(256), 0, stream>>>(
      x, wsw, qh, kh, vh);
  flash_attn_kernel<<<dim3(TT / 64, BB), dim3(128), 0, stream>>>(
      qh, kh, vh, out);
}
